// NodeSchNetWrapper_3934190044229
// MI455X (gfx1250) — matrix operand order repaired and checked by reading, compiled, not run on hardware
//
#include <hip/hip_runtime.h>
#include <hip/hip_bf16.h>
#include <cmath>

#define NTOT 1024
#define BGR  8
#define NPG  128
#define KNB  64
#define NK   (NTOT*KNB)
#define HDIM 600
#define HP   608
#define GDIM 50
#define GP   64
#define NLAY 6

typedef __attribute__((ext_vector_type(16))) __bf16 v16bf;
typedef __attribute__((ext_vector_type(8)))  __bf16 v8bf;
typedef __attribute__((ext_vector_type(8)))  float  v8f;

__device__ __forceinline__ float sspf(float x) {
    const float LN2 = 0.69314718055994531f;
    return (x > 20.f) ? (x - LN2) : (log1pf(expf(x)) - LN2);
}

// A-operand (16-bit, 16x32): lane L holds row (L&15); half h=(L>>4):
// elements 0..7 -> K = k0 + 16h + j ; elements 8..15 -> K = k0 + 16h + 8 + j
__device__ __forceinline__ v16bf load_a16(const __bf16* rowp, int k0, int half) {
    v8bf lo = *(const v8bf*)(rowp + k0 + 16 * half);
    v8bf hi = *(const v8bf*)(rowp + k0 + 16 * half + 8);
    v16bf r;
#pragma unroll
    for (int j = 0; j < 8; ++j) { r[j] = lo[j]; r[8 + j] = hi[j]; }
    return r;
}

__device__ __forceinline__ v8f wmma_bf16(v16bf a, v16bf b, v8f c) {
    return __builtin_amdgcn_wmma_f32_16x16x32_bf16(false, a, false, b, (short)0, c, false, false);
}

// ---------------------------------------------------------------------------
// Graph construction: per graph (block), per node (thread) pick the 64
// nearest in-radius neighbors. Emits nbr (global ids), d, and C = cutoff*valid.
// ---------------------------------------------------------------------------
__global__ void build_graph_kernel(const float* __restrict__ pos,
                                   int* __restrict__ nbr,
                                   float* __restrict__ dist,
                                   float* __restrict__ Cc) {
    __shared__ float px[NPG], py[NPG], pz[NPG];
    const int b = blockIdx.x, i = threadIdx.x;
    const int gi = b * NPG + i;
    px[i] = pos[gi * 3 + 0];
    py[i] = pos[gi * 3 + 1];
    pz[i] = pos[gi * 3 + 2];
    __syncthreads();
    const float xi = px[i], yi = py[i], zi = pz[i];
    unsigned mask[4] = {0u, 0u, 0u, 0u};
    mask[i >> 5] |= (1u << (i & 31));
    for (int k = 0; k < KNB; ++k) {
        float best = 1e30f; int bj = -1;
        for (int j = 0; j < NPG; ++j) {
            if (mask[j >> 5] & (1u << (j & 31))) continue;
            float dx = xi - px[j], dy = yi - py[j], dz = zi - pz[j];
            float d2 = dx * dx + dy * dy + dz * dz;
            if (d2 <= 100.0f && d2 < best) { best = d2; bj = j; }
        }
        const int e = gi * KNB + k;
        if (bj >= 0) {
            mask[bj >> 5] |= (1u << (bj & 31));
            float d = sqrtf(best);
            nbr[e]  = b * NPG + bj;
            dist[e] = d;
            Cc[e]   = 0.5f * (cosf(3.14159265358979f * d * 0.1f) + 1.0f);
        } else {
            nbr[e] = gi; dist[e] = 0.f; Cc[e] = 0.f;
        }
    }
}

// Gaussian smearing -> bf16 [NK][GP], zero padded g>=50
__global__ void smear_kernel(const float* __restrict__ dist, __bf16* __restrict__ ea) {
    const int idx = blockIdx.x * blockDim.x + threadIdx.x;
    if (idx >= NK * GP) return;
    const int e = idx >> 6, g = idx & 63;
    float v = 0.f;
    if (g < GDIM) {
        const float delta = 10.0f / 49.0f;
        const float coeff = -0.5f / (delta * delta);
        float t = dist[e] - delta * (float)g;
        v = expf(coeff * t * t);
    }
    ea[idx] = (__bf16)v;
}

// h[n][c] = embedding[z[n]][c] (pad cols zero), fp32 + bf16 copies
__global__ void embed_kernel(const int* __restrict__ z, const float* __restrict__ emb,
                             float* __restrict__ h, __bf16* __restrict__ hbf) {
    const int idx = blockIdx.x * blockDim.x + threadIdx.x;
    if (idx >= NTOT * HP) return;
    const int n = idx / HP, c = idx % HP;
    float v = (c < HDIM) ? emb[(size_t)z[n] * HDIM + c] : 0.f;
    h[idx] = v;
    hbf[idx] = (__bf16)v;
}

// Transpose+pad a [Kin][Nout] fp32 weight into bf16 Bt [Np][Kp] (row n = col of W)
__global__ void tp_kernel(const float* __restrict__ W, __bf16* __restrict__ Wt,
                          int Kin, int Nout, int Kp, int Np) {
    const int idx = blockIdx.x * blockDim.x + threadIdx.x;
    if (idx >= Np * Kp) return;
    const int n = idx / Kp, k = idx % Kp;
    float v = (n < Nout && k < Kin) ? W[(size_t)k * Nout + n] : 0.f;
    Wt[idx] = (__bf16)v;
}

// Pad biases [NLAY][HDIM] -> [NLAY][HP]
__global__ void padb_kernel(const float* __restrict__ b, float* __restrict__ bp) {
    const int idx = blockIdx.x * blockDim.x + threadIdx.x;
    if (idx >= NLAY * HP) return;
    const int l = idx / HP, c = idx % HP;
    bp[idx] = (c < HDIM) ? b[(size_t)l * HDIM + c] : 0.f;
}

// ---------------------------------------------------------------------------
// bf16 WMMA GEMM, 32x32 output tile per wave (2x2 WMMA tiles, 4 loads -> 4 wmma
// per k-step): D[M][HP] = act(A[M][HP] @ Bt^T + bias) (+addTo)
// grid = (M/32, HP/32) = (32, 19), block = 1 wave (32 threads)
// ---------------------------------------------------------------------------
__global__ void __launch_bounds__(32) gemm_kernel(const __bf16* __restrict__ A,
                                                  const __bf16* __restrict__ Bt,
                                                  const float* __restrict__ bias,
                                                  const float* __restrict__ addTo,
                                                  float* __restrict__ D,
                                                  __bf16* __restrict__ Dbf,
                                                  int act) {
    const int lane = threadIdx.x;
    const int half = lane >> 4, lc = lane & 15;
    const int m0 = blockIdx.x * 32, n0 = blockIdx.y * 32;
    const __bf16* arow0 = A  + (size_t)(m0 + lc) * HP;
    const __bf16* arow1 = arow0 + (size_t)16 * HP;
    const __bf16* brow0 = Bt + (size_t)(n0 + lc) * HP;
    const __bf16* brow1 = brow0 + (size_t)16 * HP;
    v8f c00 = {0.f,0.f,0.f,0.f,0.f,0.f,0.f,0.f};
    v8f c01 = c00, c10 = c00, c11 = c00;
    for (int k0 = 0; k0 < HP; k0 += 32) {
        v16bf a0 = load_a16(arow0, k0, half);
        v16bf a1 = load_a16(arow1, k0, half);
        v16bf b0 = *(const v16bf*)(brow0 + k0 + 16 * half);
        v16bf b1 = *(const v16bf*)(brow1 + k0 + 16 * half);
        c00 = wmma_bf16(a0, b0, c00);
        c01 = wmma_bf16(a0, b1, c01);
        c10 = wmma_bf16(a1, b0, c10);
        c11 = wmma_bf16(a1, b1, c11);
    }
    const float bv0 = bias ? bias[n0 + lc]      : 0.f;
    const float bv1 = bias ? bias[n0 + 16 + lc] : 0.f;
#pragma unroll
    for (int mi = 0; mi < 2; ++mi) {
#pragma unroll
        for (int ni = 0; ni < 2; ++ni) {
            const v8f& c = mi ? (ni ? c11 : c10) : (ni ? c01 : c00);
            const float bv = ni ? bv1 : bv0;
            const int col = n0 + ni * 16 + lc;
#pragma unroll
            for (int r = 0; r < 8; ++r) {
                const int row = m0 + mi * 16 + half * 8 + r;
                float v = c[r] + bv;
                if (act == 1) v = sspf(v);
                const size_t o = (size_t)row * HP + col;
                if (addTo) v += addTo[o];
                D[o] = v;
                if (Dbf) Dbf[o] = (__bf16)v;
            }
        }
    }
}

// ---------------------------------------------------------------------------
// Fused edge kernel: one block per node i, all 64 edges in one pass.
//   T = ssp(ea[64xGP] @ W1t + b1)            (bf16 64x608 tile in dynamic LDS)
//   Wf = (T @ W2t + b2) * C[k]               (never leaves registers)
//   msg[i][:] = sum_k Wf[k][:] * x1[nbr[k]][:]  -> bf16
// Dynamic LDS: T 77824 B + ea 8192 B + nbr 256 B + C 256 B = 86528 B
// ---------------------------------------------------------------------------
#define EDGE_SMEM (64*HP*2 + 64*GP*2 + 64*4 + 64*4)

__global__ void __launch_bounds__(128) edge_kernel(const __bf16* __restrict__ ea,
                                                   const __bf16* __restrict__ W1t,
                                                   const float* __restrict__ b1p,
                                                   const __bf16* __restrict__ W2t,
                                                   const float* __restrict__ b2p,
                                                   const float* __restrict__ x1,
                                                   const int* __restrict__ nbr,
                                                   const float* __restrict__ Cc,
                                                   __bf16* __restrict__ msg_bf) {
    extern __shared__ char smem[];
    __bf16* T    = (__bf16*)smem;                       // 64*HP bf16
    __bf16* eaL  = (__bf16*)(smem + 64 * HP * 2);       // 64*GP bf16
    int*    nbrL = (int*)  (smem + 64 * HP * 2 + 64 * GP * 2);
    float*  CL   = (float*)(smem + 64 * HP * 2 + 64 * GP * 2 + 64 * 4);

    const int i = blockIdx.x;
    const int tid = threadIdx.x;
    const int w = tid >> 5, lane = tid & 31;
    const int half = lane >> 4, lc = lane & 15;
    if (tid < KNB) { nbrL[tid] = nbr[i * KNB + tid]; CL[tid] = Cc[i * KNB + tid]; }

    // stage all 64 edge rows of ea into LDS (64*GP bf16 = 2048 dwords)
    {
        const unsigned* src = (const unsigned*)(ea + (size_t)i * KNB * GP);
        unsigned* dst = (unsigned*)eaL;
        for (int t2 = tid; t2 < KNB * GP / 2; t2 += 128) dst[t2] = src[t2];
    }
    __syncthreads();

    // GEMM1: T[64][HP] = ssp(eaL @ W1t + b1); B load hoisted over 4 m-tiles
    for (int nt = w; nt < HP / 16; nt += 4) {
        const __bf16* brow = W1t + (size_t)(nt * 16 + lc) * GP;
        v8f c[4];
#pragma unroll
        for (int mt = 0; mt < 4; ++mt) c[mt] = v8f{0.f,0.f,0.f,0.f,0.f,0.f,0.f,0.f};
#pragma unroll
        for (int k0 = 0; k0 < GP; k0 += 32) {
            v16bf b = *(const v16bf*)(brow + k0 + 16 * half);
#pragma unroll
            for (int mt = 0; mt < 4; ++mt) {
                v16bf a = load_a16(eaL + (mt * 16 + lc) * GP, k0, half);
                c[mt] = wmma_bf16(a, b, c[mt]);
            }
        }
        const float bv = b1p[nt * 16 + lc];
#pragma unroll
        for (int mt = 0; mt < 4; ++mt)
#pragma unroll
            for (int r = 0; r < 8; ++r) {
                const int row = mt * 16 + half * 8 + r;
                T[row * HP + nt * 16 + lc] = (__bf16)sspf(c[mt][r] + bv);
            }
    }
    __syncthreads();

    // GEMM2 + epilogue: B (W2t) streamed once, reused across all 4 m-tiles
    float acc[10];
#pragma unroll
    for (int t = 0; t < 10; ++t) acc[t] = 0.f;
    int t = 0;
    for (int nt = w; nt < HP / 16; nt += 4, ++t) {
        const __bf16* brow = W2t + (size_t)(nt * 16 + lc) * HP;
        const float bv = b2p[nt * 16 + lc];
        v8f c[4];
#pragma unroll
        for (int mt = 0; mt < 4; ++mt) c[mt] = v8f{0.f,0.f,0.f,0.f,0.f,0.f,0.f,0.f};
        for (int k0 = 0; k0 < HP; k0 += 32) {
            v16bf b = *(const v16bf*)(brow + k0 + 16 * half);
#pragma unroll
            for (int mt = 0; mt < 4; ++mt) {
                v16bf a = load_a16(T + (mt * 16 + lc) * HP, k0, half);
                c[mt] = wmma_bf16(a, b, c[mt]);
            }
        }
        float a2 = 0.f;
#pragma unroll
        for (int mt = 0; mt < 4; ++mt)
#pragma unroll
            for (int r = 0; r < 8; ++r) {
                const int row = mt * 16 + half * 8 + r;       // edge slot 0..63
                const float wv = (c[mt][r] + bv) * CL[row];
                a2 += wv * x1[(size_t)nbrL[row] * HP + nt * 16 + lc];
            }
        acc[t] += a2;
    }
    t = 0;
    for (int nt = w; nt < HP / 16; nt += 4, ++t) {
        float a2 = acc[t];
        a2 += __shfl_xor(a2, 16);
        if (half == 0) msg_bf[(size_t)i * HP + nt * 16 + lc] = (__bf16)a2;
    }
}

// Mean pool per graph + final linear
__global__ void pool_kernel(const float* __restrict__ h, const float* __restrict__ pw,
                            const float* __restrict__ pb, float* __restrict__ out) {
    __shared__ float pl[HDIM];
    const int b = blockIdx.x, tid = threadIdx.x;
    for (int c = tid; c < HDIM; c += 128) {
        float s = 0.f;
        for (int i2 = 0; i2 < NPG; ++i2) s += h[(size_t)(b * NPG + i2) * HP + c];
        pl[c] = s * (1.0f / NPG);
    }
    __syncthreads();
    for (int o = tid; o < HDIM; o += 128) {
        float s = pb[o];
        for (int c = 0; c < HDIM; ++c) s += pl[c] * pw[(size_t)c * HDIM + o];
        out[b * HDIM + o] = s;
    }
}

extern "C" void kernel_launch(void* const* d_in, const int* in_sizes, int n_in,
                              void* d_out, int out_size, void* d_ws, size_t ws_size,
                              hipStream_t stream) {
    (void)in_sizes; (void)n_in; (void)out_size; (void)ws_size;
    const int*   z   = (const int*)  d_in[0];
    const float* pos = (const float*)d_in[1];
    const float* emb = (const float*)d_in[2];
    const float* w1  = (const float*)d_in[3];
    const float* b1  = (const float*)d_in[4];
    const float* w2  = (const float*)d_in[5];
    const float* b2  = (const float*)d_in[6];
    const float* l1w = (const float*)d_in[7];
    const float* l2w = (const float*)d_in[8];
    const float* l2b = (const float*)d_in[9];
    const float* ilw = (const float*)d_in[10];
    const float* ilb = (const float*)d_in[11];
    const float* pw  = (const float*)d_in[12];
    const float* pb  = (const float*)d_in[13];
    float* out = (float*)d_out;

    char* ws = (char*)d_ws;
    size_t off = 0;
    auto alloc = [&](size_t bytes) -> char* {
        char* p = ws + off;
        off += (bytes + 255) & ~(size_t)255;
        return p;
    };
    int*    nbr  = (int*)   alloc((size_t)NK * 4);
    float*  dist = (float*) alloc((size_t)NK * 4);
    float*  Cc   = (float*) alloc((size_t)NK * 4);
    __bf16* ea   = (__bf16*)alloc((size_t)NK * GP * 2);
    float*  h    = (float*) alloc((size_t)NTOT * HP * 4);
    __bf16* hbf  = (__bf16*)alloc((size_t)NTOT * HP * 2);
    float*  x1   = (float*) alloc((size_t)NTOT * HP * 4);
    __bf16* msgb = (__bf16*)alloc((size_t)NTOT * HP * 2);
    float*  t1   = (float*) alloc((size_t)NTOT * HP * 4);
    __bf16* t1b  = (__bf16*)alloc((size_t)NTOT * HP * 2);
    __bf16* W1t  = (__bf16*)alloc((size_t)NLAY * HP * GP * 2);
    __bf16* W2t  = (__bf16*)alloc((size_t)NLAY * HP * HP * 2);
    __bf16* L1t  = (__bf16*)alloc((size_t)NLAY * HP * HP * 2);
    __bf16* L2t  = (__bf16*)alloc((size_t)NLAY * HP * HP * 2);
    __bf16* ILt  = (__bf16*)alloc((size_t)NLAY * HP * HP * 2);
    float*  b1p  = (float*) alloc((size_t)NLAY * HP * 4);
    float*  b2p  = (float*) alloc((size_t)NLAY * HP * 4);
    float*  l2bp = (float*) alloc((size_t)NLAY * HP * 4);
    float*  ilbp = (float*) alloc((size_t)NLAY * HP * 4);

    build_graph_kernel<<<BGR, NPG, 0, stream>>>(pos, nbr, dist, Cc);
    smear_kernel<<<(NK * GP + 255) / 256, 256, 0, stream>>>(dist, ea);
    embed_kernel<<<(NTOT * HP + 255) / 256, 256, 0, stream>>>(z, emb, h, hbf);

    for (int l = 0; l < NLAY; ++l) {
        tp_kernel<<<(HP * GP + 255) / 256, 256, 0, stream>>>(
            w1 + (size_t)l * GDIM * HDIM, W1t + (size_t)l * HP * GP, GDIM, HDIM, GP, HP);
        tp_kernel<<<(HP * HP + 255) / 256, 256, 0, stream>>>(
            w2 + (size_t)l * HDIM * HDIM, W2t + (size_t)l * HP * HP, HDIM, HDIM, HP, HP);
        tp_kernel<<<(HP * HP + 255) / 256, 256, 0, stream>>>(
            l1w + (size_t)l * HDIM * HDIM, L1t + (size_t)l * HP * HP, HDIM, HDIM, HP, HP);
        tp_kernel<<<(HP * HP + 255) / 256, 256, 0, stream>>>(
            l2w + (size_t)l * HDIM * HDIM, L2t + (size_t)l * HP * HP, HDIM, HDIM, HP, HP);
        tp_kernel<<<(HP * HP + 255) / 256, 256, 0, stream>>>(
            ilw + (size_t)l * HDIM * HDIM, ILt + (size_t)l * HP * HP, HDIM, HDIM, HP, HP);
    }
    padb_kernel<<<(NLAY * HP + 255) / 256, 256, 0, stream>>>(b1, b1p);
    padb_kernel<<<(NLAY * HP + 255) / 256, 256, 0, stream>>>(b2, b2p);
    padb_kernel<<<(NLAY * HP + 255) / 256, 256, 0, stream>>>(l2b, l2bp);
    padb_kernel<<<(NLAY * HP + 255) / 256, 256, 0, stream>>>(ilb, ilbp);

    const dim3 ggrid(NTOT / 32, HP / 32);   // (32, 19)
    for (int l = 0; l < NLAY; ++l) {
        // x1 = h @ lin1_w
        gemm_kernel<<<ggrid, 32, 0, stream>>>(hbf, L1t + (size_t)l * HP * HP,
                                              nullptr, nullptr, x1, nullptr, 0);
        // fused edge filter + aggregation -> msg (bf16)
        edge_kernel<<<NTOT, 128, EDGE_SMEM, stream>>>(
            ea, W1t + (size_t)l * HP * GP, b1p + l * HP,
            W2t + (size_t)l * HP * HP, b2p + l * HP, x1, nbr, Cc, msgb);
        // t1 = ssp(msg @ lin2 + b)
        gemm_kernel<<<ggrid, 32, 0, stream>>>(msgb, L2t + (size_t)l * HP * HP,
                                              l2bp + l * HP, nullptr, t1, t1b, 1);
        // h = h + t1 @ int_lin + b
        gemm_kernel<<<ggrid, 32, 0, stream>>>(t1b, ILt + (size_t)l * HP * HP,
                                              ilbp + l * HP, h, h, hbf, 0);
    }
    pool_kernel<<<BGR, 128, 0, stream>>>(h, pw, pb, out);
}
